// Mamba2_51823075393736
// MI455X (gfx1250) — compile-verified, matrix operand order checked by reading, not run on hardware
//
#include <hip/hip_runtime.h>
#include <math.h>

// ---------------- CDNA5 WMMA plumbing ----------------
typedef __attribute__((ext_vector_type(16))) __bf16    v16bf;
typedef __attribute__((ext_vector_type(8)))  float     v8f;
typedef __attribute__((ext_vector_type(4)))  unsigned  u32x4;

static __device__ __forceinline__ v8f wmma_bf16(v16bf a, v16bf b, v8f c) {
  return __builtin_amdgcn_wmma_f32_16x16x32_bf16(false, a, false, b, (short)0, c,
                                                 false, false);
}

// Load a 16x32 bf16 fragment from LDS (row-major, K contiguous).
// Layout per ISA 7.12.2 (16-bit A 16x32): lane r=lane&15 holds row r,
// half-lane hl=lane>>4 selects K groups: elems 0..7 -> K = 8*hl+0..7,
// elems 8..15 -> K = 16+8*hl+0..7.
static __device__ __forceinline__ v16bf frag_lds(const __bf16* base, int row0,
                                                 int stride, int kOff, int lane) {
  int r = lane & 15, hl = lane >> 4;
  const __bf16* p = base + (row0 + r) * stride + kOff + hl * 8;
  union { v16bf v; u32x4 q[2]; } u;
  u.q[0] = *(const u32x4*)(p);
  u.q[1] = *(const u32x4*)(p + 16);
  return u.v;
}

// Same fragment straight from global fp32, vectorized (4x b128 loads).
static __device__ __forceinline__ v16bf frag_gf32(const float* rowbase, int row0,
                                                  int ld, int kOff, int lane) {
  int r = lane & 15, hl = lane >> 4;
  const float* p = rowbase + (size_t)(row0 + r) * ld + kOff + hl * 8;
  float4 q0 = *(const float4*)(p);
  float4 q1 = *(const float4*)(p + 4);
  float4 q2 = *(const float4*)(p + 16);
  float4 q3 = *(const float4*)(p + 20);
  v16bf f;
  f[0]  = (__bf16)q0.x; f[1]  = (__bf16)q0.y; f[2]  = (__bf16)q0.z; f[3]  = (__bf16)q0.w;
  f[4]  = (__bf16)q1.x; f[5]  = (__bf16)q1.y; f[6]  = (__bf16)q1.z; f[7]  = (__bf16)q1.w;
  f[8]  = (__bf16)q2.x; f[9]  = (__bf16)q2.y; f[10] = (__bf16)q2.z; f[11] = (__bf16)q2.w;
  f[12] = (__bf16)q3.x; f[13] = (__bf16)q3.y; f[14] = (__bf16)q3.z; f[15] = (__bf16)q3.w;
  return f;
}

static __device__ __forceinline__ v8f vzero8() {
  v8f z = {0.f, 0.f, 0.f, 0.f, 0.f, 0.f, 0.f, 0.f};
  return z;
}

// ---------------- problem constants ----------------
#define BATCH   2
#define SEQLEN  2048
#define DMODEL  1024
#define DINNER  2048
#define NHEADS  32
#define HEADDIM 64
#define NSTATE  64
#define CHUNKQ  128
#define NCHUNK  16
#define DINPROJ 4256
#define CONV_TL 8

// ================= Kernel 1/6: NT GEMM (C = A * W^T), bf16 WMMA =================
// Block tile 128x128, 8 waves of 32x64 (8 WMMAs per wave per K-step).
// Register-pipelined: next K-tile loads overlap current-tile WMMAs.
__global__ __launch_bounds__(256) void gemm_nt_bf16(const float* __restrict__ A,
                                                    const float* __restrict__ W,
                                                    float* __restrict__ C,
                                                    int M, int N, int K) {
  __shared__ __bf16 sA[128 * 40];  // 128 rows x 32 K, stride 40 (bank spread)
  __shared__ __bf16 sB[128 * 40];  // 128 rows (N) x 32 K
  const int tid = threadIdx.x, lane = tid & 31, wave = tid >> 5;
  const int hl = lane >> 4, l15 = lane & 15;
  const int m0 = blockIdx.y * 128, n0 = blockIdx.x * 128;
  const int wm = wave & 3, wn = wave >> 2;  // 4x2 wave grid, 32x64 each

  v8f acc[2][4];
#pragma unroll
  for (int i = 0; i < 2; ++i)
#pragma unroll
    for (int j = 0; j < 4; ++j) acc[i][j] = vzero8();

  const int rt = tid >> 3;       // 0..31
  const int ct = (tid & 7) * 4;  // 0..28 step 4

  float4 va[4], vb[4];
  // prologue: load K-tile 0 into registers (all loads issue before any use)
#pragma unroll
  for (int rr = 0; rr < 4; ++rr)
    va[rr] = *(const float4*)(A + (size_t)(m0 + rt + rr * 32) * K + ct);
#pragma unroll
  for (int rr = 0; rr < 4; ++rr) {
    int n = n0 + rt + rr * 32;
    float4 v; v.x = v.y = v.z = v.w = 0.f;
    if (n < N) v = *(const float4*)(W + (size_t)n * K + ct);
    vb[rr] = v;
  }

  for (int k0 = 0; k0 < K; k0 += 32) {
    // commit current register tile to LDS (fp32 -> bf16)
#pragma unroll
    for (int rr = 0; rr < 4; ++rr) {
      __bf16* d = &sA[(rt + rr * 32) * 40 + ct];
      d[0] = (__bf16)va[rr].x; d[1] = (__bf16)va[rr].y;
      d[2] = (__bf16)va[rr].z; d[3] = (__bf16)va[rr].w;
      __bf16* e = &sB[(rt + rr * 32) * 40 + ct];
      e[0] = (__bf16)vb[rr].x; e[1] = (__bf16)vb[rr].y;
      e[2] = (__bf16)vb[rr].z; e[3] = (__bf16)vb[rr].w;
    }
    __syncthreads();

    // prefetch next K-tile into registers; waits land at next commit,
    // so the global latency overlaps the WMMAs below.
    const int k1 = k0 + 32;
    if (k1 < K) {
#pragma unroll
      for (int rr = 0; rr < 4; ++rr)
        va[rr] = *(const float4*)(A + (size_t)(m0 + rt + rr * 32) * K + k1 + ct);
#pragma unroll
      for (int rr = 0; rr < 4; ++rr) {
        int n = n0 + rt + rr * 32;
        float4 v; v.x = v.y = v.z = v.w = 0.f;
        if (n < N) v = *(const float4*)(W + (size_t)n * K + k1 + ct);
        vb[rr] = v;
      }
    }
    // L2 prefetch two K-steps ahead (global_prefetch_b8 on gfx1250)
    if (k0 + 64 < K) {
      __builtin_prefetch(A + (size_t)(m0 + rt) * K + k0 + 64 + ct, 0, 1);
      if (n0 + rt < N)
        __builtin_prefetch(W + (size_t)(n0 + rt) * K + k0 + 64 + ct, 0, 1);
    }

    v16bf af[2], bf[4];
#pragma unroll
    for (int i = 0; i < 2; ++i) af[i] = frag_lds(sA, wm * 32 + i * 16, 40, 0, lane);
#pragma unroll
    for (int j = 0; j < 4; ++j) bf[j] = frag_lds(sB, wn * 64 + j * 16, 40, 0, lane);
#pragma unroll
    for (int i = 0; i < 2; ++i)
#pragma unroll
      for (int j = 0; j < 4; ++j) acc[i][j] = wmma_bf16(af[i], bf[j], acc[i][j]);
    __syncthreads();
  }

#pragma unroll
  for (int i = 0; i < 2; ++i)
#pragma unroll
    for (int j = 0; j < 4; ++j)
#pragma unroll
      for (int e = 0; e < 8; ++e) {
        int m = m0 + wm * 32 + i * 16 + e + 8 * hl;
        int n = n0 + wn * 64 + j * 16 + l15;
        if (n < N) C[(size_t)m * N + n] = acc[i][j][e];
      }
}

// ================= Kernel 2/6: causal depthwise conv4 + bias + split =================
static __device__ __forceinline__ void route_store(
    float acc, long bl, int c, const float* __restrict__ dt_bias,
    const float* __restrict__ A_log, float* __restrict__ z, float* __restrict__ x,
    float* __restrict__ Bg, float* __restrict__ Cg, float* __restrict__ dt,
    float* __restrict__ a) {
  if (c < DINNER) {
    z[bl * DINNER + c] = acc;
  } else if (c < 2 * DINNER) {
    x[bl * DINNER + (c - DINNER)] = acc;
  } else if (c < 2 * DINNER + NSTATE) {
    Bg[bl * NSTATE + (c - 2 * DINNER)] = acc;
  } else if (c < 2 * DINNER + 2 * NSTATE) {
    Cg[bl * NSTATE + (c - 2 * DINNER - NSTATE)] = acc;
  } else {
    int hh = c - (2 * DINNER + 2 * NSTATE);
    float raw = acc + dt_bias[hh];
    float sp = (raw > 20.f) ? raw : log1pf(expf(raw));
    dt[bl * NHEADS + hh] = sp;
    a[bl * NHEADS + hh] = sp * A_log[hh];
  }
}

// Sliding window: each thread produces CONV_TL consecutive l for one channel
// (cuts zx re-reads from 4x to ~1.4x; loads stay coalesced in c).
__global__ __launch_bounds__(256) void conv_split(
    const float* __restrict__ zx, const float* __restrict__ conv_w,
    const float* __restrict__ conv_b, const float* __restrict__ dt_bias,
    const float* __restrict__ A_log, float* __restrict__ z, float* __restrict__ x,
    float* __restrict__ Bg, float* __restrict__ Cg, float* __restrict__ dt,
    float* __restrict__ a, long total) {
  long idx = (long)blockIdx.x * blockDim.x + threadIdx.x;
  if (idx >= total) return;
  int c = (int)(idx % DINPROJ);
  long bl0 = (idx / DINPROJ) * CONV_TL;
  int l0 = (int)(bl0 % SEQLEN);
  float w0 = conv_w[c * 4 + 0], w1 = conv_w[c * 4 + 1];
  float w2 = conv_w[c * 4 + 2], w3 = conv_w[c * 4 + 3];
  float bias = conv_b[c];
  const float* col = zx + c;
  float p0 = (l0 >= 3) ? col[(size_t)(bl0 - 3) * DINPROJ] : 0.f;
  float p1 = (l0 >= 2) ? col[(size_t)(bl0 - 2) * DINPROJ] : 0.f;
  float p2 = (l0 >= 1) ? col[(size_t)(bl0 - 1) * DINPROJ] : 0.f;
#pragma unroll
  for (int j = 0; j < CONV_TL; ++j) {
    long bl = bl0 + j;
    float cur = col[(size_t)bl * DINPROJ];
    float acc = bias + w0 * p0 + w1 * p1 + w2 * p2 + w3 * cur;
    p0 = p1; p1 = p2; p2 = cur;
    route_store(acc, bl, c, dt_bias, A_log, z, x, Bg, Cg, dt, a);
  }
}

// ================= Kernel 3/6: per-chunk inclusive cumsum of a -> s =================
__global__ __launch_bounds__(128) void chunk_cumsum(const float* __restrict__ a,
                                                    float* __restrict__ s) {
  __shared__ float buf[2][CHUNKQ];
  int h = blockIdx.x & 31;
  int bc = blockIdx.x >> 5;          // b*16 + chunk
  int q = threadIdx.x;
  long row = (long)bc * CHUNKQ + q;
  int src = 0;
  buf[0][q] = a[row * NHEADS + h];
  __syncthreads();
#pragma unroll
  for (int off = 1; off < CHUNKQ; off <<= 1) {
    float v = buf[src][q];
    if (q >= off) v += buf[src][q - off];
    buf[1 - src][q] = v;
    src = 1 - src;
    __syncthreads();
  }
  s[row * NHEADS + h] = buf[src][q];
}

// ================= Kernel 4/6: intra-chunk (CB -> mask -> y) + chunk states =================
__global__ __launch_bounds__(128) void intra_chunk(
    const float* __restrict__ xbuf, const float* __restrict__ Bg,
    const float* __restrict__ Cg, const float* __restrict__ dtb,
    const float* __restrict__ sb, const float* __restrict__ Dparam,
    float* __restrict__ ybuf, float* __restrict__ dhbuf) {
  const int bc = blockIdx.x;   // b*16 + chunk
  const int h = blockIdx.y;
  const int tid = threadIdx.x, lane = tid & 31, wave = tid >> 5;
  const int hl = lane >> 4, l15 = lane & 15;
  const size_t bl0 = (size_t)bc * CHUNKQ;

  __shared__ float ldsS[CHUNKQ], ldsDT[CHUNKQ], ldsW[CHUNKQ];
  __shared__ __bf16 ldsB[CHUNKQ * 72];   // B rows q, K=n (state)
  __shared__ __bf16 ldsXT[64 * 136];     // x^T : rows p, cols q
  __shared__ __bf16 ldsM[64 * 136];      // half of masked M; reused for (B*w)^T

  ldsS[tid]  = sb[(bl0 + tid) * NHEADS + h];
  ldsDT[tid] = dtb[(bl0 + tid) * NHEADS + h];
  __syncthreads();
  const float sLast = ldsS[CHUNKQ - 1];
  ldsW[tid] = ldsDT[tid] * expf(sLast - ldsS[tid]);
  for (int e = tid; e < CHUNKQ * NSTATE; e += 128) {
    int q = e >> 6, n = e & 63;
    ldsB[q * 72 + n] = (__bf16)Bg[(bl0 + q) * NSTATE + n];
  }
  for (int e = tid; e < CHUNKQ * HEADDIM; e += 128) {
    int q = e >> 6, p = e & 63;
    ldsXT[p * 136 + q] = (__bf16)xbuf[(bl0 + q) * DINNER + h * HEADDIM + p];
  }
  __syncthreads();

  const float* Crow = Cg + bl0 * NSTATE;

  for (int half = 0; half < 2; ++half) {
    const int rowOff = half * 64;
    const int i0 = rowOff + wave * 16;

    // -------- stage 1: CB = C * B^T for this 16-row band --------
    v8f acc[8];
#pragma unroll
    for (int j = 0; j < 8; ++j) acc[j] = vzero8();
#pragma unroll
    for (int ks = 0; ks < NSTATE; ks += 32) {
      v16bf af = frag_gf32(Crow, i0, NSTATE, ks, lane);
#pragma unroll
      for (int j = 0; j < 8; ++j) {
        v16bf bf = frag_lds(ldsB, j * 16, 72, ks, lane);
        acc[j] = wmma_bf16(af, bf, acc[j]);
      }
    }
    // mask: M[i,j] = CB * exp(s_i - s_j) * dt_j  (j <= i), write bf16 to LDS
#pragma unroll
    for (int j = 0; j < 8; ++j)
#pragma unroll
      for (int e = 0; e < 8; ++e) {
        int i = i0 + e + 8 * hl;
        int jj = j * 16 + l15;
        float v = 0.f;
        if (jj <= i) v = acc[j][e] * expf(ldsS[i] - ldsS[jj]) * ldsDT[jj];
        ldsM[(i - rowOff) * 136 + jj] = (__bf16)v;
      }
    __syncthreads();

    // -------- stage 2: y = M * x  (+ D*x) --------
    v8f yacc[4];
#pragma unroll
    for (int t = 0; t < 4; ++t) yacc[t] = vzero8();
#pragma unroll
    for (int ks = 0; ks < CHUNKQ; ks += 32) {
      v16bf af = frag_lds(ldsM, wave * 16, 136, ks, lane);
#pragma unroll
      for (int t = 0; t < 4; ++t) {
        v16bf bf = frag_lds(ldsXT, t * 16, 136, ks, lane);
        yacc[t] = wmma_bf16(af, bf, yacc[t]);
      }
    }
#pragma unroll
    for (int t = 0; t < 4; ++t)
#pragma unroll
      for (int e = 0; e < 8; ++e) {
        int i = rowOff + wave * 16 + e + 8 * hl;
        int p = t * 16 + l15;
        size_t gi = (bl0 + i) * DINNER + h * HEADDIM + p;
        float xv = xbuf[gi];
        ybuf[gi] = yacc[t][e] + Dparam[h * HEADDIM + p] * xv;
      }
    __syncthreads();
  }

  // -------- chunk state: dh[n,p] = sum_q B[q,n]*w_q*x[q,p] --------
  for (int e = tid; e < CHUNKQ * NSTATE; e += 128) {
    int q = e >> 6, n = e & 63;
    ldsM[n * 136 + q] = (__bf16)(Bg[(bl0 + q) * NSTATE + n] * ldsW[q]);
  }
  __syncthreads();
  v8f dacc[4];
#pragma unroll
  for (int t = 0; t < 4; ++t) dacc[t] = vzero8();
#pragma unroll
  for (int ks = 0; ks < CHUNKQ; ks += 32) {
    v16bf af = frag_lds(ldsM, wave * 16, 136, ks, lane);
#pragma unroll
    for (int t = 0; t < 4; ++t) {
      v16bf bf = frag_lds(ldsXT, t * 16, 136, ks, lane);
      dacc[t] = wmma_bf16(af, bf, dacc[t]);
    }
  }
  size_t dhBase = ((size_t)bc * NHEADS + h) * (NSTATE * HEADDIM);
#pragma unroll
  for (int t = 0; t < 4; ++t)
#pragma unroll
    for (int e = 0; e < 8; ++e) {
      int n = wave * 16 + e + 8 * hl;
      int p = t * 16 + l15;
      dhbuf[dhBase + n * HEADDIM + p] = dacc[t][e];
    }
}

// ================= Kernel 5/6: sequential inter-chunk recurrence =================
__global__ __launch_bounds__(128) void inter_scan(const float* __restrict__ Cg,
                                                  const float* __restrict__ sb,
                                                  const float* __restrict__ dhbuf,
                                                  float* __restrict__ ybuf) {
  const int b = blockIdx.x >> 5, h = blockIdx.x & 31;
  const int tid = threadIdx.x, lane = tid & 31, wave = tid >> 5;
  const int hl = lane >> 4, l15 = lane & 15;

  __shared__ float hS[NSTATE * HEADDIM];  // fp32 master state h[n][p]
  __shared__ __bf16 hT[64 * 72];          // h^T bf16: rows p, cols n
  __shared__ float ldsS[CHUNKQ];

  for (int e = tid; e < NSTATE * HEADDIM; e += 128) hS[e] = 0.f;
  for (int e = tid; e < 64 * 72; e += 128) hT[e] = (__bf16)0.f;
  __syncthreads();

  for (int c = 0; c < NCHUNK; ++c) {
    size_t bl0 = (size_t)b * SEQLEN + (size_t)c * CHUNKQ;
    ldsS[tid] = sb[(bl0 + tid) * NHEADS + h];
    __syncthreads();

    if (c > 0) {
      const int i0 = wave * 32;
      v8f acc[2][4];
#pragma unroll
      for (int rt = 0; rt < 2; ++rt)
#pragma unroll
        for (int t = 0; t < 4; ++t) acc[rt][t] = vzero8();
#pragma unroll
      for (int ks = 0; ks < NSTATE; ks += 32) {
        v16bf a0 = frag_gf32(Cg + bl0 * NSTATE, i0, NSTATE, ks, lane);
        v16bf a1 = frag_gf32(Cg + bl0 * NSTATE, i0 + 16, NSTATE, ks, lane);
#pragma unroll
        for (int t = 0; t < 4; ++t) {
          v16bf bf = frag_lds(hT, t * 16, 72, ks, lane);
          acc[0][t] = wmma_bf16(a0, bf, acc[0][t]);
          acc[1][t] = wmma_bf16(a1, bf, acc[1][t]);
        }
      }
#pragma unroll
      for (int rt = 0; rt < 2; ++rt)
#pragma unroll
        for (int t = 0; t < 4; ++t)
#pragma unroll
          for (int e = 0; e < 8; ++e) {
            int i = i0 + rt * 16 + e + 8 * hl;
            int p = t * 16 + l15;
            size_t gi = (bl0 + i) * DINNER + h * HEADDIM + p;
            ybuf[gi] += acc[rt][t][e] * expf(ldsS[i]);
          }
    }
    __syncthreads();

    float decay = expf(ldsS[CHUNKQ - 1]);
    size_t dhBase = ((size_t)(b * NCHUNK + c) * NHEADS + h) * (NSTATE * HEADDIM);
    for (int e = tid; e < NSTATE * HEADDIM; e += 128)
      hS[e] = decay * hS[e] + dhbuf[dhBase + e];
    __syncthreads();
    for (int e = tid; e < NSTATE * HEADDIM; e += 128) {
      int n = e >> 6, p = e & 63;
      hT[p * 72 + n] = (__bf16)hS[e];
    }
    __syncthreads();
  }
}

// ================= Kernel 6/6: RMSNorm + SiLU gate =================
__global__ __launch_bounds__(256) void norm_gate(const float* __restrict__ y,
                                                 const float* __restrict__ z,
                                                 const float* __restrict__ norm_w,
                                                 float* __restrict__ gated) {
  __shared__ float red[256];
  size_t base = (size_t)blockIdx.x * DINNER;
  float ss = 0.f;
  for (int d = threadIdx.x; d < DINNER; d += 256) {
    float v = y[base + d];
    ss += v * v;
  }
  red[threadIdx.x] = ss;
  __syncthreads();
  for (int o = 128; o > 0; o >>= 1) {
    if ((int)threadIdx.x < o) red[threadIdx.x] += red[threadIdx.x + o];
    __syncthreads();
  }
  float rinv = rsqrtf(red[0] / (float)DINNER + 1.1920929e-7f);
  for (int d = threadIdx.x; d < DINNER; d += 256) {
    float yv = y[base + d] * rinv * norm_w[d];
    float zv = z[base + d];
    gated[base + d] = yv * (zv / (1.f + expf(-zv)));
  }
}

// ================= host-side launcher =================
extern "C" void kernel_launch(void* const* d_in, const int* in_sizes, int n_in,
                              void* d_out, int out_size, void* d_ws, size_t ws_size,
                              hipStream_t stream) {
  (void)in_sizes; (void)n_in; (void)out_size; (void)ws_size;
  const float* u          = (const float*)d_in[0];
  const float* in_proj_w  = (const float*)d_in[1];
  const float* conv_w     = (const float*)d_in[2];
  const float* conv_b     = (const float*)d_in[3];
  const float* A_log      = (const float*)d_in[4];
  const float* Dparam     = (const float*)d_in[5];
  const float* dt_bias    = (const float*)d_in[6];
  const float* norm_w     = (const float*)d_in[7];
  const float* out_proj_w = (const float*)d_in[8];

  const long BL = (long)BATCH * SEQLEN;           // 4096 tokens
  float* w = (float*)d_ws;
  // workspace layout (floats)
  size_t ZX = 0;                                  // [BL, 4256] pre-conv (reused later)
  size_t Z  = ZX + (size_t)BL * DINPROJ;          // [BL, 2048]
  size_t X  = Z + (size_t)BL * DINNER;            // [BL, 2048]
  size_t BG = X + (size_t)BL * DINNER;            // [BL, 64]
  size_t CGo = BG + (size_t)BL * NSTATE;          // [BL, 64]
  size_t DT = CGo + (size_t)BL * NSTATE;          // [BL, 32]
  size_t AA = DT + (size_t)BL * NHEADS;           // [BL, 32]
  size_t SS = AA + (size_t)BL * NHEADS;           // [BL, 32]
  size_t DH = SS + (size_t)BL * NHEADS;           // [B,nc,H,64,64]
  // reuse the (large) ZX region once conv is done:
  float* ybuf  = w + 0;                           // [BL, 2048]
  float* gated = w + (size_t)BL * DINNER;         // [BL, 2048] (fits in ZX region)

  // 1) in_proj GEMM: zx = u @ in_proj_w^T   [4096 x 4256]
  gemm_nt_bf16<<<dim3((DINPROJ + 127) / 128, (int)(BL / 128)), 256, 0, stream>>>(
      u, in_proj_w, w + ZX, (int)BL, DINPROJ, DMODEL);

  // 2) conv + bias + split + softplus(dt), sliding window of 8
  long total = (BL / CONV_TL) * DINPROJ;
  conv_split<<<(int)((total + 255) / 256), 256, 0, stream>>>(
      w + ZX, conv_w, conv_b, dt_bias, A_log,
      w + Z, w + X, w + BG, w + CGo, w + DT, w + AA, total);

  // 3) per-chunk cumsum: s
  chunk_cumsum<<<BATCH * NCHUNK * NHEADS, 128, 0, stream>>>(w + AA, w + SS);

  // 4) intra-chunk WMMA (overwrites zx region with y, safe now)
  intra_chunk<<<dim3(BATCH * NCHUNK, NHEADS), 128, 0, stream>>>(
      w + X, w + BG, w + CGo, w + DT, w + SS, Dparam, ybuf, w + DH);

  // 5) inter-chunk sequential scan (adds y_inter into ybuf)
  inter_scan<<<BATCH * NHEADS, 128, 0, stream>>>(w + CGo, w + SS, w + DH, ybuf);

  // 6) RMSNorm + gate
  norm_gate<<<(int)BL, 256, 0, stream>>>(ybuf, w + Z, norm_w, gated);

  // 7) out_proj GEMM: out = gated @ out_proj_w^T  [4096 x 1024]
  gemm_nt_bf16<<<dim3(DMODEL / 128, (int)(BL / 128)), 256, 0, stream>>>(
      gated, out_proj_w, (float*)d_out, (int)BL, DMODEL, DINNER);
}
